// HyenaOperator_81303730913667
// MI455X (gfx1250) — matrix operand order repaired and checked by reading, not compiled, not run on hardware
//
#include <hip/hip_runtime.h>
#include <hip/hip_bf16.h>
#include <math.h>

// ---------------------------------------------------------------------------
// HyenaOperator for MI455X (gfx1250, wave32, WMMA).
//
// x (B=4,L=4096,D=1024) f32 ; W1 (1024,4096); b1(4096); W2 (4096,6); b2(6)
// out (B,L,D) f32.
//
// GEMM1 (x@W1) = 137 GFLOP dominates -> v_wmma_f32_16x16x32_bf16.
// W1^T bf16 = 8 MB -> resident in 192 MB L2; HBM traffic stays ~137 MB floor.
// ---------------------------------------------------------------------------

typedef __attribute__((ext_vector_type(16))) __bf16 v16bf;
typedef __attribute__((ext_vector_type(8)))  __bf16 v8bf;
typedef __attribute__((ext_vector_type(8)))  float  v8f;

#define D_IN   1024
#define H_DIM  4096
#define N_OUT  6
#define LDS_PITCH (D_IN + 8)   // bf16 elems; +16B/row to spread LDS banks

__device__ __forceinline__ float gelu_f(float x) {
    return 0.5f * x * (1.0f + erff(x * 0.70710678118654752440f));
}

__device__ __forceinline__ v8f wmma_bf16(const v16bf& a, const v16bf& b, const v8f& c) {
    return __builtin_amdgcn_wmma_f32_16x16x32_bf16(false, a, false, b, (short)0, c,
                                                   false, false);
}

// ---------------------------------------------------------------------------
// Kernel 1: W1 (1024 x 4096 f32) -> W1T (4096 x 1024 bf16)
// ---------------------------------------------------------------------------
__global__ void w1_to_bf16T(const float* __restrict__ w1, __bf16* __restrict__ w1t) {
    __shared__ float tile[32][33];
    const int n0 = blockIdx.x * 32;
    const int k0 = blockIdx.y * 32;
    const int tx = threadIdx.x;   // 0..31
    const int ty = threadIdx.y;   // 0..7
    #pragma unroll
    for (int i = ty; i < 32; i += 8)
        tile[i][tx] = w1[(size_t)(k0 + i) * H_DIM + n0 + tx];   // tile[k][n]
    __syncthreads();
    #pragma unroll
    for (int i = ty; i < 32; i += 8)
        w1t[(size_t)(n0 + i) * D_IN + k0 + tx] = (__bf16)tile[tx][i];
}

// ---------------------------------------------------------------------------
// Kernel 2: params = GELU(x@W1 + b1) @ W2 + b2, fused.
// Block = 128 threads (4 waves), one 16-token M-tile in LDS (bf16).
// Each wave owns 64 N-tiles, processed as 32 *pairs*: both tiles of a pair
// share the A fragment and run two independent WMMA chains, K double-buffered.
// An opaque zero offset makes the A pointer look loop-variant so the compiler
// cannot hoist all 32 K-steps of LDS A loads out of the s-loop (which spilled
// 1 KB/lane to scratch in the previous revision).
//
// WMMA fragment layouts used here (wave32; A/B k order contiguous, C/D as in ISA 7.12.2):
//   A 16x32 bf16 : lane -> M=lane&15 ; elem e -> K = e + (lane&16)
//   B 32x16 bf16 : lane -> N=lane&15 ; elem e -> K = e + (lane&16)
//   C/D f32      : VGPR i, lanes 0..15 -> (M=i, N=lane); 16..31 -> (M=i+8)
// ---------------------------------------------------------------------------
__global__ void paramgen(const float* __restrict__ x,      // (B*L, 1024)
                         const __bf16* __restrict__ w1t,   // (4096, 1024)
                         const float* __restrict__ b1,     // (4096)
                         const float* __restrict__ w2,     // (4096, 6)
                         const float* __restrict__ b2,     // (6)
                         float* __restrict__ params) {     // (B*L, 6)
    extern __shared__ char smem[];
    __bf16* xs    = (__bf16*)smem;                          // [16][LDS_PITCH]
    float*  pTile = (float*)(smem + 16 * LDS_PITCH * 2);    // [16][6]

    const int tid     = threadIdx.x;     // 0..127
    const int wave    = tid >> 5;        // 0..3
    const int lane    = tid & 31;
    const int lane16  = lane & 15;
    const int tokBase = blockIdx.x * 16;

    // ---- stage 16 tokens x 1024 ch of x into LDS as bf16 ----
    for (int i = tid; i < 16 * (D_IN / 4); i += 128) {
        const int row  = i >> 8;
        const int col4 = (i & 255) * 4;
        const float4 v = *(const float4*)(x + (size_t)(tokBase + row) * D_IN + col4);
        __bf16* dst = xs + row * LDS_PITCH + col4;
        dst[0] = (__bf16)v.x; dst[1] = (__bf16)v.y;
        dst[2] = (__bf16)v.z; dst[3] = (__bf16)v.w;
    }
    for (int i = tid; i < 16 * N_OUT; i += 128) pTile[i] = 0.0f;
    __syncthreads();

    float pAcc[8][N_OUT];
    #pragma unroll
    for (int i = 0; i < 8; ++i)
        #pragma unroll
        for (int o = 0; o < N_OUT; ++o) pAcc[i][o] = 0.0f;

    const __bf16* arowBase = xs + lane16 * LDS_PITCH + (lane & 16);

    // wave owns nt = wave + 4*t, t = 0..63 ; process pairs (2s, 2s+1)
    for (int s = 0; s < 32; ++s) {
        // opaque zero: keeps arow "loop-variant" -> no cross-s hoist of ds loads
        int sdep = 0;
        asm volatile("" : "+v"(sdep));
        const __bf16* arow = arowBase + sdep;

        const int nG0 = (wave + 4 * (2 * s))     * 16 + lane16;
        const int nG1 = (wave + 4 * (2 * s + 1)) * 16 + lane16;
        const __bf16* brow0 = w1t + (size_t)nG0 * D_IN + (lane & 16);
        const __bf16* brow1 = w1t + (size_t)nG1 * D_IN + (lane & 16);

        v8f c0 = {0.f,0.f,0.f,0.f,0.f,0.f,0.f,0.f};
        v8f c1 = {0.f,0.f,0.f,0.f,0.f,0.f,0.f,0.f};

        // prologue: fragments for kk = 0
        v8bf  a0 = *(const v8bf*)(arow);
        v8bf  a1 = *(const v8bf*)(arow + 8);
        v16bf b0 = *(const v16bf*)(brow0);
        v16bf b1 = *(const v16bf*)(brow1);

        #pragma unroll 4
        for (int kk = 0; kk < D_IN - 32; kk += 32) {
            // preload next K-step while this step's WMMAs execute
            const int kn = kk + 32;
            v8bf  na0 = *(const v8bf*)(arow + kn);
            v8bf  na1 = *(const v8bf*)(arow + kn + 8);
            v16bf nb0 = *(const v16bf*)(brow0 + kn);
            v16bf nb1 = *(const v16bf*)(brow1 + kn);

            v16bf a;
            #pragma unroll
            for (int e = 0; e < 8; ++e) { a[e] = a0[e]; a[e + 8] = a1[e]; }
            c0 = wmma_bf16(a, b0, c0);
            c1 = wmma_bf16(a, b1, c1);

            a0 = na0; a1 = na1; b0 = nb0; b1 = nb1;
        }
        {   // peeled last K-step
            v16bf a;
            #pragma unroll
            for (int e = 0; e < 8; ++e) { a[e] = a0[e]; a[e + 8] = a1[e]; }
            c0 = wmma_bf16(a, b0, c0);
            c1 = wmma_bf16(a, b1, c1);
        }

        // fused epilogue for both tiles: +b1, GELU, x W2 row (col n = lane16)
        const float bias0 = b1[nG0], bias1 = b1[nG1];
        float w2r0[N_OUT], w2r1[N_OUT];
        #pragma unroll
        for (int o = 0; o < N_OUT; ++o) {
            w2r0[o] = w2[(size_t)nG0 * N_OUT + o];
            w2r1[o] = w2[(size_t)nG1 * N_OUT + o];
        }
        #pragma unroll
        for (int i = 0; i < 8; ++i) {
            const float g0 = gelu_f(c0[i] + bias0);
            const float g1 = gelu_f(c1[i] + bias1);
            #pragma unroll
            for (int o = 0; o < N_OUT; ++o) {
                pAcc[i][o] = fmaf(g0, w2r0[o], pAcc[i][o]);
                pAcc[i][o] = fmaf(g1, w2r1[o], pAcc[i][o]);
            }
        }
    }

    // reduce over the 16 hidden-columns held by the lanes of each half-wave
    #pragma unroll
    for (int m = 1; m < 16; m <<= 1)
        #pragma unroll
        for (int i = 0; i < 8; ++i)
            #pragma unroll
            for (int o = 0; o < N_OUT; ++o)
                pAcc[i][o] += __shfl_xor(pAcc[i][o], m, 32);

    if (lane16 == 0) {  // lane 0 -> rows 0..7, lane 16 -> rows 8..15
        const int rBase = (lane >> 4) * 8;
        #pragma unroll
        for (int i = 0; i < 8; ++i)
            #pragma unroll
            for (int o = 0; o < N_OUT; ++o)
                atomicAdd(&pTile[(rBase + i) * N_OUT + o], pAcc[i][o]); // ds_add_f32
    }
    __syncthreads();

    for (int i = tid; i < 16 * N_OUT; i += 128) {
        const int tok = i / N_OUT, o = i % N_OUT;
        params[(size_t)(tokBase + tok) * N_OUT + o] = pTile[i] + b2[o];
    }
}

// ---------------------------------------------------------------------------
// Kernel 3: fused double causal dynamic conv + GELU (memory-bound).
// out[l] = GELU( k1[0]*g[l-2] + k1[1]*g[l-1] + k1[2]*g[l] ),
//   g[l] = GELU( k0[0]*x[l-2] + k0[1]*x[l-1] + k0[2]*x[l] ),  zero left-pad.
// L chunked with a 4-token recompute halo; 256 threads x float4 = 1024 ch.
// ---------------------------------------------------------------------------
#define CHUNK 64

__global__ void dynconv(const float* __restrict__ x,
                        const float* __restrict__ params,
                        float* __restrict__ out, int L) {
    const int b  = blockIdx.y;
    const int l0 = blockIdx.x * CHUNK;
    const int c4 = threadIdx.x * 4;
    const float4* xin = (const float4*)(x   + (size_t)b * L * D_IN + c4);
    float4*       po  = (float4*)      (out + (size_t)b * L * D_IN + c4);

    float4 xm1 = {0,0,0,0}, xm2 = {0,0,0,0};
    float4 gm1 = {0,0,0,0}, gm2 = {0,0,0,0};

    for (int l = l0 - 4; l < l0 + CHUNK; ++l) {
        float4 xv = {0,0,0,0}, g = {0,0,0,0};
        if (l >= 0) {
            xv = xin[(size_t)l * (D_IN / 4)];
            const float* p = params + ((size_t)b * L + l) * N_OUT; // broadcast
            const float k00 = p[0], k01 = p[1], k02 = p[2];
            const float k10 = p[3], k11 = p[4], k12 = p[5];
            g.x = gelu_f(fmaf(k00, xm2.x, fmaf(k01, xm1.x, k02 * xv.x)));
            g.y = gelu_f(fmaf(k00, xm2.y, fmaf(k01, xm1.y, k02 * xv.y)));
            g.z = gelu_f(fmaf(k00, xm2.z, fmaf(k01, xm1.z, k02 * xv.z)));
            g.w = gelu_f(fmaf(k00, xm2.w, fmaf(k01, xm1.w, k02 * xv.w)));
            if (l >= l0) {
                float4 ov;
                ov.x = gelu_f(fmaf(k10, gm2.x, fmaf(k11, gm1.x, k12 * g.x)));
                ov.y = gelu_f(fmaf(k10, gm2.y, fmaf(k11, gm1.y, k12 * g.y)));
                ov.z = gelu_f(fmaf(k10, gm2.z, fmaf(k11, gm1.z, k12 * g.z)));
                ov.w = gelu_f(fmaf(k10, gm2.w, fmaf(k11, gm1.w, k12 * g.w)));
                po[(size_t)l * (D_IN / 4)] = ov;
            }
        }
        xm2 = xm1; xm1 = xv;
        gm2 = gm1; gm1 = g;
    }
}

// ---------------------------------------------------------------------------
extern "C" void kernel_launch(void* const* d_in, const int* in_sizes, int n_in,
                              void* d_out, int out_size, void* d_ws, size_t ws_size,
                              hipStream_t stream) {
    const float* x  = (const float*)d_in[0];
    const float* W1 = (const float*)d_in[1];
    const float* b1 = (const float*)d_in[2];
    const float* W2 = (const float*)d_in[3];
    const float* b2 = (const float*)d_in[4];
    float* out = (float*)d_out;

    const int B = 4, L = 4096;
    const int nTok = B * L;

    __bf16* w1t = (__bf16*)d_ws;                                    // 8 MB
    float*  prm = (float*)((char*)d_ws + (size_t)H_DIM * D_IN * 2); // 384 KB

    w1_to_bf16T<<<dim3(H_DIM / 32, D_IN / 32), dim3(32, 8), 0, stream>>>(W1, w1t);

    const size_t smem = (size_t)16 * LDS_PITCH * 2 + 16 * N_OUT * 4; // ~33.4 KB
    paramgen<<<dim3(nTok / 16), dim3(128), smem, stream>>>(x, w1t, b1, W2, b2, prm);

    dynconv<<<dim3(L / CHUNK, B), dim3(256), 0, stream>>>(x, prm, out, L);
}
